// ConCareLayer_59193239273724
// MI455X (gfx1250) — compile-verified, matrix operand order checked by reading, not run on hardware
//
#include <hip/hip_runtime.h>
#include <math.h>

#define B_SZ 128
#define T_SZ 64
#define D_SZ 64
#define H_SZ 128
#define RB   32   // batch rows per GRU block

typedef __attribute__((ext_vector_type(16))) _Float16     v16h;
typedef __attribute__((ext_vector_type(8)))  float        v8f;
typedef __attribute__((ext_vector_type(4)))  unsigned int u32x4;

union FragU { v16h v; u32x4 q[2]; _Float16 h[16]; };

// A/B fragment load when the K axis is contiguous in memory (row-major [rows][K]).
// ISA 7.12.2: lanes 0-15 -> rows, lanes 16-31 -> same rows, second K half-group.
// element e<8  : K = kbase + g*8 + e
// element e>=8 : K = kbase + 16 + g*8 + (e-8)
__device__ __forceinline__ v16h load_frag_k(const _Float16* base, int row0, int ld,
                                            int kbase, int lane) {
  const int r = row0 + (lane & 15);
  const int g = lane >> 4;
  const _Float16* p = base + r * ld + kbase + g * 8;
  FragU u;
  u.q[0] = *(const u32x4*)(p);
  u.q[1] = *(const u32x4*)(p + 16);
  return u.v;
}

// B fragment load when source is row-major [K][N] (K strided): per-element gather.
__device__ __forceinline__ v16h load_frag_rowmajor(const _Float16* base, int n0, int ld,
                                                   int kbase, int lane) {
  const int n = n0 + (lane & 15);
  const int g = lane >> 4;
  FragU u;
#pragma unroll
  for (int i = 0; i < 16; ++i) {
    const int K = kbase + g * 8 + (i < 8 ? i : i + 8);
    u.h[i] = base[K * ld + n];
  }
  return u.v;
}

__device__ __forceinline__ v8f wmma_f16(v16h a, v16h b, v8f c) {
  return __builtin_amdgcn_wmma_f32_16x16x32_f16(false, a, false, b, (short)0, c,
                                                false, false);
}

__device__ __forceinline__ float sigmoidf_(float x) { return 1.0f / (1.0f + expf(-x)); }

// ---------------------------------------------------------------------------
// Prep kernels: f32 -> f16 convert / transpose into workspace
// ---------------------------------------------------------------------------
__global__ void k_cvt_f16(const float* __restrict__ s, _Float16* __restrict__ d, int n) {
  int i = blockIdx.x * blockDim.x + threadIdx.x;
  if (i < n) d[i] = (_Float16)s[i];
}

__global__ void k_transpose_f16(const float* __restrict__ s, _Float16* __restrict__ d,
                                int K, int N) {
  int i = blockIdx.x * blockDim.x + threadIdx.x;
  if (i < K * N) {
    int k = i / N, n = i - k * N;
    d[n * K + k] = (_Float16)s[i];   // dst is [N][K], K contiguous
  }
}

// ---------------------------------------------------------------------------
// GRU recurrence: grid (d, batch-chunk). Per block: 32 batch rows of feature d.
// gh[b,k] = sum_h h[b,h] * Whh[d,k,h]  -> GEMM M=32, N=384, K=128 per step.
// ---------------------------------------------------------------------------
__global__ __launch_bounds__(256, 1)
void gru_kernel(const float* __restrict__ x, const int* __restrict__ mask,
                const float* __restrict__ Wih, const float* __restrict__ bih,
                const float* __restrict__ bhh, const _Float16* __restrict__ whh16,
                float* __restrict__ embws)
{
  __shared__ float    gh[RB * 388];        // 49664 B
  __shared__ _Float16 h16[RB * 136];       // 8704 B (rows 16B aligned)
  __shared__ float    xs[RB];
  __shared__ float    wihS[384], bihS[384], bhhS[384];
  __shared__ int      idxS[RB];

  const int tid  = threadIdx.x;
  const int lane = tid & 31;
  const int w    = tid >> 5;
  const int d    = blockIdx.x;
  const int bg0  = blockIdx.y * RB;

  for (int k = tid; k < 384; k += 256) {
    wihS[k] = Wih[d * 384 + k];
    bihS[k] = bih[d * 384 + k];
    bhhS[k] = bhh[d * 384 + k];
  }
  for (int i = tid; i < RB * 136; i += 256) h16[i] = (_Float16)0.0f;
  if (tid < RB) {
    int len = 0;
    for (int t = 0; t < T_SZ; ++t) len += mask[(bg0 + tid) * T_SZ + t];
    int ix = len - 1;
    if (ix < 0) ix = 0;
    if (ix > T_SZ - 1) ix = T_SZ - 1;
    idxS[tid] = ix;
  }

  // Whh[d] B-fragments resident in VGPRs for the whole time loop.
  // wave w owns n-tiles {3w, 3w+1, 3w+2} of 24 (N=384).
  const _Float16* whhd = whh16 + (size_t)d * 384 * 128;
  v16h bf[3][4];
#pragma unroll
  for (int nt = 0; nt < 3; ++nt)
#pragma unroll
    for (int kf = 0; kf < 4; ++kf)
      bf[nt][kf] = load_frag_k(whhd, (w * 3 + nt) * 16, 128, kf * 32, lane);

  __syncthreads();

  for (int t = 0; t < T_SZ; ++t) {
    if (tid < RB) xs[tid] = x[((bg0 + tid) * T_SZ + t) * D_SZ + d];
    __syncthreads();

    // gh = h @ Whh^T
    for (int mt = 0; mt < RB / 16; ++mt) {
      v16h af[4];
#pragma unroll
      for (int kf = 0; kf < 4; ++kf)
        af[kf] = load_frag_k(h16, mt * 16, 136, kf * 32, lane);
#pragma unroll
      for (int nt = 0; nt < 3; ++nt) {
        v8f acc = {0.f, 0.f, 0.f, 0.f, 0.f, 0.f, 0.f, 0.f};
#pragma unroll
        for (int kf = 0; kf < 4; ++kf)
          acc = wmma_f16(af[kf], bf[nt][kf], acc);
        const int col = (w * 3 + nt) * 16 + (lane & 15);
        const int rb  = mt * 16 + ((lane >> 4) << 3);
#pragma unroll
        for (int r = 0; r < 8; ++r)
          gh[(rb + r) * 388 + col] = acc[r];
      }
    }
    __syncthreads();

    // GRU cell elementwise (gate order r, z, n)
    for (int i = tid; i < RB * H_SZ; i += 256) {
      const int   b  = i >> 7;
      const int   hh = i & 127;
      const float xv = xs[b];
      const float gr = xv * wihS[hh]       + bihS[hh]       + gh[b * 388 + hh]       + bhhS[hh];
      const float gz = xv * wihS[128 + hh] + bihS[128 + hh] + gh[b * 388 + 128 + hh] + bhhS[128 + hh];
      const float r  = sigmoidf_(gr);
      const float z  = sigmoidf_(gz);
      const float gn = xv * wihS[256 + hh] + bihS[256 + hh] +
                       r * (gh[b * 388 + 256 + hh] + bhhS[256 + hh]);
      const float n  = tanhf(gn);
      const float hp = (float)h16[b * 136 + hh];
      const float hn = (1.0f - z) * n + z * hp;
      h16[b * 136 + hh] = (_Float16)hn;
      if (t == idxS[b])
        embws[((size_t)(bg0 + b) * D_SZ + d) * H_SZ + hh] = hn;
    }
    __syncthreads();
  }
}

// ---------------------------------------------------------------------------
// Post stage: MHA + FFN + LayerNorms + final attention. One block per batch b.
// Dynamic LDS layout (bytes), with phase-based region reuse:
// ---------------------------------------------------------------------------
#define P_EMB   0        // emb f32 [64][128] (later: ffn16 [64][264] f16, 33792B)
#define P_A16   32768    // emb f16 [64][136]
#define P_Q16   50176    // q16 [64][136] (later: h2f32 [64][128], spans Q16+K16)
#define P_K16   67584    // k16 [64][136]
#define P_V16   84992    // v16 [64][136] (later: h216 [64][136])
#define P_SC    102400   // scores f32 [4][64][64]
#define P_ATT   167936   // attn f16 [4][64][64] (later: fvf32 [64][128])
#define P_O16   200704   // o16 [64][136]
#define P_H1F   218112   // h1 f32 [64][128] (later: fkf32 [64][128])
#define P_H116  250880   // h1 f16 [64][136]
#define P_MISC  268288   // fq[128] f32 + e/a[64] f32
#define P_TOT   269312

__global__ __launch_bounds__(256, 1)
void post_kernel(const float* __restrict__ embws,
                 const _Float16* __restrict__ wqt, const float* __restrict__ bq,
                 const _Float16* __restrict__ wkt, const float* __restrict__ bk,
                 const _Float16* __restrict__ wvt, const float* __restrict__ bv,
                 const _Float16* __restrict__ wot, const float* __restrict__ bo,
                 const _Float16* __restrict__ w1t, const float* __restrict__ b1,
                 const _Float16* __restrict__ w2t, const float* __restrict__ b2,
                 const float* __restrict__ g1, const float* __restrict__ be1,
                 const float* __restrict__ g2, const float* __restrict__ be2,
                 const float* __restrict__ Fq, const float* __restrict__ fbq,
                 const _Float16* __restrict__ fkt, const float* __restrict__ fbk,
                 const _Float16* __restrict__ fvt, const float* __restrict__ fbv,
                 const float* __restrict__ Fout, const float* __restrict__ fbout,
                 float* __restrict__ out)
{
  extern __shared__ __align__(16) unsigned char smem[];
  float*    embf  = (float*)(smem + P_EMB);
  _Float16* a16   = (_Float16*)(smem + P_A16);
  _Float16* q16   = (_Float16*)(smem + P_Q16);
  _Float16* k16   = (_Float16*)(smem + P_K16);
  _Float16* v16   = (_Float16*)(smem + P_V16);
  float*    sc    = (float*)(smem + P_SC);
  _Float16* att   = (_Float16*)(smem + P_ATT);
  _Float16* o16   = (_Float16*)(smem + P_O16);
  float*    h1f   = (float*)(smem + P_H1F);
  _Float16* h116  = (_Float16*)(smem + P_H116);
  _Float16* ffn16 = (_Float16*)(smem + P_EMB);   // overlay: emb dead after h1
  float*    h2f   = (float*)(smem + P_Q16);      // overlay: q16/k16 dead
  _Float16* h216  = (_Float16*)(smem + P_V16);   // overlay: v16 dead
  float*    fkf   = (float*)(smem + P_H1F);      // overlay: h1 dead after h2
  float*    fvf   = (float*)(smem + P_ATT);      // overlay: attn dead
  float*    fqv   = (float*)(smem + P_MISC);     // [128]
  float*    ea    = (float*)(smem + P_MISC + 512); // [64]

  const int tid = threadIdx.x, lane = tid & 31, w = tid >> 5;
  const int b   = blockIdx.x;
  const float* emb_b = embws + (size_t)b * 64 * 128;

  // A: stage emb (f32 + f16)
  for (int i = tid; i < 64 * 128; i += 256) {
    float v = emb_b[i];
    int r = i >> 7, c = i & 127;
    embf[i] = v;
    a16[r * 136 + c] = (_Float16)v;
  }
  __syncthreads();

  // B: q,k,v = emb @ W + b  (M=64,N=128,K=128); wave w owns n-tile w
  {
    const _Float16* wt[3]  = {wqt, wkt, wvt};
    const float*    bs[3]  = {bq, bk, bv};
    _Float16*       dst[3] = {q16, k16, v16};
    for (int mt = 0; mt < 4; ++mt) {
      v16h af[4];
#pragma unroll
      for (int kf = 0; kf < 4; ++kf)
        af[kf] = load_frag_k(a16, mt * 16, 136, kf * 32, lane);
#pragma unroll
      for (int mat = 0; mat < 3; ++mat) {
        v8f acc = {0.f, 0.f, 0.f, 0.f, 0.f, 0.f, 0.f, 0.f};
#pragma unroll
        for (int kf = 0; kf < 4; ++kf)
          acc = wmma_f16(af[kf], load_frag_k(wt[mat], w * 16, 128, kf * 32, lane), acc);
        const int   n    = w * 16 + (lane & 15);
        const float bias = bs[mat][n];
        const int   rb   = mt * 16 + ((lane >> 4) << 3);
#pragma unroll
        for (int r = 0; r < 8; ++r)
          dst[mat][(rb + r) * 136 + n] = (_Float16)(acc[r] + bias);
      }
    }
  }
  __syncthreads();

  // C: scores = q @ k^T / sqrt(dk)   (4 heads, dk=32; 64 tiles, K=32 -> 1 WMMA)
  for (int j = 0; j < 8; ++j) {
    const int tI = w + 8 * j;
    const int hd = tI >> 4, rem = tI & 15, mt = rem >> 2, nt = rem & 3;
    v16h aq = load_frag_k(q16, mt * 16, 136, hd * 32, lane);
    v16h bk_ = load_frag_k(k16, nt * 16, 136, hd * 32, lane);  // k^T: K contiguous
    v8f acc = {0.f, 0.f, 0.f, 0.f, 0.f, 0.f, 0.f, 0.f};
    acc = wmma_f16(aq, bk_, acc);
    const int n  = nt * 16 + (lane & 15);
    const int rb = mt * 16 + ((lane >> 4) << 3);
#pragma unroll
    for (int r = 0; r < 8; ++r)
      sc[hd * 4096 + (rb + r) * 64 + n] = acc[r] * 0.17677669529663687f;
  }
  __syncthreads();

  // D: softmax over 256 rows (4 heads * 64), one row per thread
  {
    float* row = sc + tid * 64;
    float mx = row[0];
    for (int i = 1; i < 64; ++i) mx = fmaxf(mx, row[i]);
    float s = 0.f;
    for (int i = 0; i < 64; ++i) s += expf(row[i] - mx);
    float inv = 1.0f / s;
    for (int i = 0; i < 64; ++i)
      att[tid * 64 + i] = (_Float16)(expf(row[i] - mx) * inv);
  }
  __syncthreads();

  // E: o = attn @ v  (per head M=64,N=32,K=64; 32 tiles)
  for (int j = 0; j < 4; ++j) {
    const int tI = w + 8 * j;
    const int hd = tI >> 3, rem = tI & 7, mt = rem >> 1, nt = rem & 1;
    v8f acc = {0.f, 0.f, 0.f, 0.f, 0.f, 0.f, 0.f, 0.f};
#pragma unroll
    for (int kf = 0; kf < 2; ++kf) {
      v16h aa = load_frag_k(att + hd * 4096, mt * 16, 64, kf * 32, lane);
      v16h bb = load_frag_rowmajor(v16, hd * 32 + nt * 16, 136, kf * 32, lane);
      acc = wmma_f16(aa, bb, acc);
    }
    const int n  = hd * 32 + nt * 16 + (lane & 15);
    const int rb = mt * 16 + ((lane >> 4) << 3);
#pragma unroll
    for (int r = 0; r < 8; ++r)
      o16[(rb + r) * 136 + n] = (_Float16)acc[r];
  }
  __syncthreads();

  // F: h1pre = o @ Wo + bo + emb
  for (int mt = 0; mt < 4; ++mt) {
    v16h af[4];
#pragma unroll
    for (int kf = 0; kf < 4; ++kf)
      af[kf] = load_frag_k(o16, mt * 16, 136, kf * 32, lane);
    v8f acc = {0.f, 0.f, 0.f, 0.f, 0.f, 0.f, 0.f, 0.f};
#pragma unroll
    for (int kf = 0; kf < 4; ++kf)
      acc = wmma_f16(af[kf], load_frag_k(wot, w * 16, 128, kf * 32, lane), acc);
    const int   n    = w * 16 + (lane & 15);
    const float bias = bo[n];
    const int   rb   = mt * 16 + ((lane >> 4) << 3);
#pragma unroll
    for (int r = 0; r < 8; ++r)
      h1f[(rb + r) * 128 + n] = acc[r] + bias + embf[(rb + r) * 128 + n];
  }
  __syncthreads();

  // G: LayerNorm1 (one row per thread)
  if (tid < 64) {
    float* row = h1f + tid * 128;
    float m = 0.f;
    for (int c = 0; c < 128; ++c) m += row[c];
    m *= (1.0f / 128.0f);
    float v = 0.f;
    for (int c = 0; c < 128; ++c) { float dd = row[c] - m; v += dd * dd; }
    v *= (1.0f / 128.0f);
    float inv = rsqrtf(v + 1e-5f);
    for (int c = 0; c < 128; ++c) {
      float val = (row[c] - m) * inv * g1[c] + be1[c];
      row[c] = val;
      h116[tid * 136 + c] = (_Float16)val;
    }
  }
  __syncthreads();

  // H: ffn1 = relu(h1 @ W1 + b1)  (N=256; wave w owns n-tiles w, w+8)
  for (int jj = 0; jj < 2; ++jj) {
    const int nt = w + 8 * jj;
    for (int mt = 0; mt < 4; ++mt) {
      v8f acc = {0.f, 0.f, 0.f, 0.f, 0.f, 0.f, 0.f, 0.f};
#pragma unroll
      for (int kf = 0; kf < 4; ++kf)
        acc = wmma_f16(load_frag_k(h116, mt * 16, 136, kf * 32, lane),
                       load_frag_k(w1t, nt * 16, 128, kf * 32, lane), acc);
      const int   n    = nt * 16 + (lane & 15);
      const float bias = b1[n];
      const int   rb   = mt * 16 + ((lane >> 4) << 3);
#pragma unroll
      for (int r = 0; r < 8; ++r)
        ffn16[(rb + r) * 264 + n] = (_Float16)fmaxf(acc[r] + bias, 0.f);
    }
  }
  __syncthreads();

  // I: h2pre = ffn1 @ W2 + b2 + h1  (K=256)
  for (int mt = 0; mt < 4; ++mt) {
    v8f acc = {0.f, 0.f, 0.f, 0.f, 0.f, 0.f, 0.f, 0.f};
#pragma unroll
    for (int kf = 0; kf < 8; ++kf)
      acc = wmma_f16(load_frag_k(ffn16, mt * 16, 264, kf * 32, lane),
                     load_frag_k(w2t, w * 16, 256, kf * 32, lane), acc);
    const int   n    = w * 16 + (lane & 15);
    const float bias = b2[n];
    const int   rb   = mt * 16 + ((lane >> 4) << 3);
#pragma unroll
    for (int r = 0; r < 8; ++r)
      h2f[(rb + r) * 128 + n] = acc[r] + bias + h1f[(rb + r) * 128 + n];
  }
  __syncthreads();

  // J: LayerNorm2
  if (tid < 64) {
    float* row = h2f + tid * 128;
    float m = 0.f;
    for (int c = 0; c < 128; ++c) m += row[c];
    m *= (1.0f / 128.0f);
    float v = 0.f;
    for (int c = 0; c < 128; ++c) { float dd = row[c] - m; v += dd * dd; }
    v *= (1.0f / 128.0f);
    float inv = rsqrtf(v + 1e-5f);
    for (int c = 0; c < 128; ++c) {
      float val = (row[c] - m) * inv * g2[c] + be2[c];
      row[c] = val;
      h216[tid * 136 + c] = (_Float16)val;
    }
  }
  __syncthreads();

  // K: fq = h2[last] @ Fq + fbq  (single row, VALU)
  if (tid < 128) {
    float s = fbq[tid];
    const float* h2last = h2f + 63 * 128;
    for (int h = 0; h < 128; ++h) s += h2last[h] * Fq[h * 128 + tid];
    fqv[tid] = s;
  }
  __syncthreads();

  // L: fk = h2 @ Fk + fbk ; fv = h2 @ Fv + fbv
  {
    const _Float16* ft[2] = {fkt, fvt};
    const float*    fb[2] = {fbk, fbv};
    float*          fd[2] = {fkf, fvf};
    for (int mt = 0; mt < 4; ++mt) {
      v16h af[4];
#pragma unroll
      for (int kf = 0; kf < 4; ++kf)
        af[kf] = load_frag_k(h216, mt * 16, 136, kf * 32, lane);
#pragma unroll
      for (int mat = 0; mat < 2; ++mat) {
        v8f acc = {0.f, 0.f, 0.f, 0.f, 0.f, 0.f, 0.f, 0.f};
#pragma unroll
        for (int kf = 0; kf < 4; ++kf)
          acc = wmma_f16(af[kf], load_frag_k(ft[mat], w * 16, 128, kf * 32, lane), acc);
        const int   n    = w * 16 + (lane & 15);
        const float bias = fb[mat][n];
        const int   rb   = mt * 16 + ((lane >> 4) << 3);
#pragma unroll
        for (int r = 0; r < 8; ++r)
          fd[mat][(rb + r) * 128 + n] = acc[r] + bias;
      }
    }
  }
  __syncthreads();

  // M: e[m] = tanh(fq + fk[m]) . Fout + fbout
  if (tid < 64) {
    float s = fbout[0];
    for (int h = 0; h < 128; ++h) s += tanhf(fqv[h] + fkf[tid * 128 + h]) * Fout[h];
    ea[tid] = s;
  }
  __syncthreads();

  // N: softmax over 64 (serial, trivial)
  if (tid == 0) {
    float mx = ea[0];
    for (int i = 1; i < 64; ++i) mx = fmaxf(mx, ea[i]);
    float s = 0.f;
    for (int i = 0; i < 64; ++i) s += expf(ea[i] - mx);
    float inv = 1.0f / s;
    for (int i = 0; i < 64; ++i) ea[i] = expf(ea[i] - mx) * inv;
  }
  __syncthreads();

  // O: out[b,h] = sum_m a[m] * fv[m,h]
  if (tid < 128) {
    float s = 0.f;
    for (int m = 0; m < 64; ++m) s += ea[m] * fvf[m * 128 + tid];
    out[b * 128 + tid] = s;
  }
}

// ---------------------------------------------------------------------------
// Workspace layout (bytes)
// ---------------------------------------------------------------------------
static const size_t OFF_EMB   = 0;                                       // 4,194,304
static const size_t OFF_WHH16 = OFF_EMB   + (size_t)B_SZ * D_SZ * H_SZ * 4;
static const size_t OFF_WQT   = OFF_WHH16 + (size_t)D_SZ * 384 * 128 * 2;
static const size_t OFF_WKT   = OFF_WQT   + 128 * 128 * 2;
static const size_t OFF_WVT   = OFF_WKT   + 128 * 128 * 2;
static const size_t OFF_WOT   = OFF_WVT   + 128 * 128 * 2;
static const size_t OFF_W1T   = OFF_WOT   + 128 * 128 * 2;
static const size_t OFF_W2T   = OFF_W1T   + 256 * 128 * 2;
static const size_t OFF_FKT   = OFF_W2T   + 128 * 256 * 2;
static const size_t OFF_FVT   = OFF_FKT   + 128 * 128 * 2;

extern "C" void kernel_launch(void* const* d_in, const int* in_sizes, int n_in,
                              void* d_out, int out_size, void* d_ws, size_t ws_size,
                              hipStream_t stream) {
  (void)in_sizes; (void)n_in; (void)out_size; (void)ws_size;
  const float* x    = (const float*)d_in[0];
  const int*   mask = (const int*)d_in[1];
  const float* Wih  = (const float*)d_in[2];
  const float* Whh  = (const float*)d_in[3];
  const float* bih  = (const float*)d_in[4];
  const float* bhh  = (const float*)d_in[5];
  const float* Wq = (const float*)d_in[6];  const float* bq = (const float*)d_in[7];
  const float* Wk = (const float*)d_in[8];  const float* bk = (const float*)d_in[9];
  const float* Wv = (const float*)d_in[10]; const float* bv = (const float*)d_in[11];
  const float* Wo = (const float*)d_in[12]; const float* bo = (const float*)d_in[13];
  const float* W1 = (const float*)d_in[14]; const float* b1 = (const float*)d_in[15];
  const float* W2 = (const float*)d_in[16]; const float* b2 = (const float*)d_in[17];
  const float* g1 = (const float*)d_in[18]; const float* be1 = (const float*)d_in[19];
  const float* g2 = (const float*)d_in[20]; const float* be2 = (const float*)d_in[21];
  const float* Fq = (const float*)d_in[22]; const float* fbq = (const float*)d_in[23];
  const float* Fk = (const float*)d_in[24]; const float* fbk = (const float*)d_in[25];
  const float* Fv = (const float*)d_in[26]; const float* fbv = (const float*)d_in[27];
  const float* Fout = (const float*)d_in[28];
  const float* fbout = (const float*)d_in[29];
  float* out = (float*)d_out;

  unsigned char* ws = (unsigned char*)d_ws;
  float*    embws = (float*)(ws + OFF_EMB);
  _Float16* whh16 = (_Float16*)(ws + OFF_WHH16);
  _Float16* wqt = (_Float16*)(ws + OFF_WQT);
  _Float16* wkt = (_Float16*)(ws + OFF_WKT);
  _Float16* wvt = (_Float16*)(ws + OFF_WVT);
  _Float16* wot = (_Float16*)(ws + OFF_WOT);
  _Float16* w1t = (_Float16*)(ws + OFF_W1T);
  _Float16* w2t = (_Float16*)(ws + OFF_W2T);
  _Float16* fkt = (_Float16*)(ws + OFF_FKT);
  _Float16* fvt = (_Float16*)(ws + OFF_FVT);

  // Prep: weight conversion / transposition
  const int nWhh = D_SZ * 384 * 128;
  k_cvt_f16<<<(nWhh + 255) / 256, 256, 0, stream>>>(Whh, whh16, nWhh);
  k_transpose_f16<<<64, 256, 0, stream>>>(Wq, wqt, 128, 128);
  k_transpose_f16<<<64, 256, 0, stream>>>(Wk, wkt, 128, 128);
  k_transpose_f16<<<64, 256, 0, stream>>>(Wv, wvt, 128, 128);
  k_transpose_f16<<<64, 256, 0, stream>>>(Wo, wot, 128, 128);
  k_transpose_f16<<<128, 256, 0, stream>>>(W1, w1t, 128, 256);
  k_transpose_f16<<<128, 256, 0, stream>>>(W2, w2t, 256, 128);
  k_transpose_f16<<<64, 256, 0, stream>>>(Fk, fkt, 128, 128);
  k_transpose_f16<<<64, 256, 0, stream>>>(Fv, fvt, 128, 128);

  // GRU recurrence: 64 features x 4 batch chunks
  gru_kernel<<<dim3(D_SZ, B_SZ / RB), 256, 0, stream>>>(x, mask, Wih, bih, bhh,
                                                        whh16, embws);

  // MHA + FFN + final attention: one block per batch element
  post_kernel<<<B_SZ, 256, P_TOT, stream>>>(embws,
                                            wqt, bq, wkt, bk, wvt, bv, wot, bo,
                                            w1t, b1, w2t, b2,
                                            g1, be1, g2, be2,
                                            Fq, fbq, fkt, fbk, fvt, fbv,
                                            Fout, fbout, out);
}
